// CrossAttentionOutLayer_57191784513822
// MI455X (gfx1250) — compile-verified, matrix operand order checked by reading, not run on hardware
//
#include <hip/hip_runtime.h>
#include <hip/hip_bf16.h>

// CDNA5 / gfx1250 wave32 WMMA GEMM pipeline for CrossAttentionOutLayer.
//
// Math reduction: mean over heads of (q*SCALE + rel_bias)@k^T collapses to a
// single 512-wide contraction:
//   out[b,n,m] = sum_c ( (A.Wq^T)[b,n,c]*SCALE/H + (bq[c]*SCALE+rel[c])/H )
//                * (P.Wk^T + bk)[b,m,c]
// => GEMM1: Qp(bf16) = rna @ Wq^T   (alpha=SCALE/H, colbias=(bq*SCALE+rel)/H)
//    GEMM2: Kp(bf16) = prot @ Wk^T  (+bk)
//    GEMM3: out(f32) = Qp @ Kp^T per batch (8 x 1024x1024x512)
//
// All three are "C = A * Bt^T" with the contraction dim contiguous in both
// operands -> v_wmma_f32_16x16x32_bf16 fragments are two contiguous 16B LDS
// chunks per lane (K = half*8..+7 and +16), i.e. two ds_load_b128 each.
//
// CDNA5 data movement:
//  * f32 inputs (projections): double-buffered LDS; next tile's global loads
//    issue into VGPRs before the WMMA block, convert+ds_store after it.
//  * bf16 inputs (logits): GLOBAL_LOAD_ASYNC_TO_LDS_B128 (ASYNCcnt-tracked,
//    no VGPR data) double-buffered; s_wait_asynccnt 0 before the barrier.
//
// Workspace: Qp 8192*512*2 B + Kp 8192*512*2 B = 16 MiB in d_ws.

typedef __attribute__((ext_vector_type(16))) __bf16        v16bf;
typedef __attribute__((ext_vector_type(4)))  __bf16        v4bf;
typedef __attribute__((ext_vector_type(8)))  float         v8f;
typedef __attribute__((ext_vector_type(4)))  float         v4f;
typedef __attribute__((ext_vector_type(4)))  unsigned int  v4u;
typedef __attribute__((ext_vector_type(8)))  unsigned int  v8u;

#define BM 128
#define BN 128
#define BKT 64
#define LDT 72   // padded LDS row length in bf16 elems (64 + 8) -> 144B rows

// Per-lane 16B global->LDS async copy; lds_off is the wave-relative LDS byte
// address (low 32 bits of a generic pointer into __shared__, per ISA §10.2).
__device__ __forceinline__ void async_copy_b128(unsigned lds_off, const void* g)
{
    asm volatile("global_load_async_to_lds_b128 %0, %1, off"
                 :: "v"(lds_off), "v"(g)
                 : "memory");
}
__device__ __forceinline__ void wait_async0()
{
    asm volatile("s_wait_asynccnt 0x0" ::: "memory");
}

// C[row, col] = alpha * sum_k A[row,k]*Bt[col,k] + s0*bias0[col] + s1*bias1[col]
template <typename TIN, typename TOUT>
__global__ __launch_bounds__(256)
void wmma_gemm_nt(const TIN* __restrict__ A,  int lda, long long strideA,
                  const TIN* __restrict__ Bt, int ldb, long long strideB,
                  TOUT* __restrict__ C,       int ldc, long long strideC,
                  int Kdim,
                  const float* __restrict__ bias0, float s0,
                  const float* __restrict__ bias1, float s1,
                  float alpha)
{
    constexpr bool ASYNCIN = (sizeof(TIN) == 2);   // bf16 in -> async-to-LDS

    __shared__ __bf16 sA[2][BM * LDT];
    __shared__ __bf16 sB[2][BN * LDT];

    A  += (size_t)blockIdx.z * strideA;
    Bt += (size_t)blockIdx.z * strideB;
    C  += (size_t)blockIdx.z * strideC;

    const int tid  = threadIdx.x;
    const int wid  = tid >> 5;            // 8 waves
    const int lane = tid & 31;
    const int l16  = lane & 15;
    const int hi16 = lane >> 4;           // 16-lane half

    const int block_m = blockIdx.x * BM;
    const int block_n = blockIdx.y * BN;

    // wave tile 32(M) x 64(N); waves arranged 4(M) x 2(N)
    const int wm = (wid >> 1) * 32;
    const int wn = (wid & 1) * 64;

    v8f acc[2][4];
#pragma unroll
    for (int i = 0; i < 2; ++i)
#pragma unroll
        for (int j = 0; j < 4; ++j)
#pragma unroll
            for (int r = 0; r < 8; ++r) acc[i][j][r] = 0.0f;

    // f32 staging map: 256 threads cover 128 rows x 16 float4 chunks
    const int s_c4 = tid & 15;
    const int s_r0 = tid >> 4;
    v4f rA[8], rB[8];

    // ---- staging helpers ------------------------------------------------
    auto issue_async = [&](int b, int kt) {
        if constexpr (ASYNCIN) {
            // 128 rows x 8 chunks of 16B per operand; 4 chunks/thread each
#pragma unroll
            for (int i = 0; i < 4; ++i) {
                const int chunk = tid + i * 256;        // 0..1023
                const int r = chunk >> 3;
                const int c = chunk & 7;
                const unsigned la =
                    (unsigned)(unsigned long long)(uintptr_t)&sA[b][r * LDT + c * 8];
                const unsigned lb =
                    (unsigned)(unsigned long long)(uintptr_t)&sB[b][r * LDT + c * 8];
                async_copy_b128(la, (const char*)(A + (size_t)(block_m + r) * lda + kt) + c * 16);
                async_copy_b128(lb, (const char*)(Bt + (size_t)(block_n + r) * ldb + kt) + c * 16);
            }
        }
    };
    auto load_regs = [&](int kt) {
        if constexpr (!ASYNCIN) {
#pragma unroll
            for (int i = 0; i < 8; ++i)
                rA[i] = *(const v4f*)(A + (size_t)(block_m + s_r0 + i * 16) * lda + kt + s_c4 * 4);
#pragma unroll
            for (int i = 0; i < 8; ++i)
                rB[i] = *(const v4f*)(Bt + (size_t)(block_n + s_r0 + i * 16) * ldb + kt + s_c4 * 4);
        }
    };
    auto store_tile = [&](int b) {
        if constexpr (!ASYNCIN) {
#pragma unroll
            for (int i = 0; i < 8; ++i) {
                v4bf p;
                p[0] = (__bf16)rA[i][0]; p[1] = (__bf16)rA[i][1];
                p[2] = (__bf16)rA[i][2]; p[3] = (__bf16)rA[i][3];
                *(v4bf*)(&sA[b][(s_r0 + i * 16) * LDT + s_c4 * 4]) = p;
            }
#pragma unroll
            for (int i = 0; i < 8; ++i) {
                v4bf p;
                p[0] = (__bf16)rB[i][0]; p[1] = (__bf16)rB[i][1];
                p[2] = (__bf16)rB[i][2]; p[3] = (__bf16)rB[i][3];
                *(v4bf*)(&sB[b][(s_r0 + i * 16) * LDT + s_c4 * 4]) = p;
            }
        }
    };

    // ---- prologue: fill buffer 0 with tile kt=0 -------------------------
    if constexpr (ASYNCIN) {
        issue_async(0, 0);
        wait_async0();
    } else {
        load_regs(0);
        store_tile(0);
    }
    __syncthreads();

    // ---- ping-pong main loop -------------------------------------------
    int buf = 0;
    for (int kt = 0; kt < Kdim; kt += BKT) {
        const bool has_next = (kt + BKT) < Kdim;

        if (has_next) {
            if constexpr (ASYNCIN) issue_async(buf ^ 1, kt + BKT);
            else                   load_regs(kt + BKT);
        }

        // compute on current buffer
#pragma unroll
        for (int kb = 0; kb < BKT; kb += 32) {
            v16bf fa[2];
#pragma unroll
            for (int i = 0; i < 2; ++i) {
                const __bf16* base = &sA[buf][(wm + i * 16 + l16) * LDT + kb + hi16 * 8];
                const v4u lo = *(const v4u*)base;          // K = hi16*8 .. +7
                const v4u hh = *(const v4u*)(base + 16);   // K = 16+hi16*8 .. +7
                fa[i] = __builtin_bit_cast(
                    v16bf, __builtin_shufflevector(lo, hh, 0, 1, 2, 3, 4, 5, 6, 7));
            }
            v16bf fb[4];
#pragma unroll
            for (int j = 0; j < 4; ++j) {
                const __bf16* base = &sB[buf][(wn + j * 16 + l16) * LDT + kb + hi16 * 8];
                const v4u lo = *(const v4u*)base;
                const v4u hh = *(const v4u*)(base + 16);
                fb[j] = __builtin_bit_cast(
                    v16bf, __builtin_shufflevector(lo, hh, 0, 1, 2, 3, 4, 5, 6, 7));
            }
#pragma unroll
            for (int i = 0; i < 2; ++i)
#pragma unroll
                for (int j = 0; j < 4; ++j)
                    acc[i][j] = __builtin_amdgcn_wmma_f32_16x16x32_bf16(
                        false, fa[i], false, fb[j], (short)0, acc[i][j],
                        false, false);
        }

        if (has_next) {
            if constexpr (ASYNCIN) wait_async0();   // next buffer's copies done
            else                   store_tile(buf ^ 1);
            __syncthreads();                        // publish next buffer
        }
        buf ^= 1;
    }

    // ---- epilogue: VGPR r holds row (hi16*8 + r), lane%16 = col ---------
#pragma unroll
    for (int i = 0; i < 2; ++i) {
#pragma unroll
        for (int j = 0; j < 4; ++j) {
            const int col = block_n + wn + j * 16 + l16;
            float cb = 0.0f;
            if (bias0) cb += s0 * bias0[col];
            if (bias1) cb += s1 * bias1[col];
#pragma unroll
            for (int r = 0; r < 8; ++r) {
                const int row = block_m + wm + i * 16 + hi16 * 8 + r;
                const float v = acc[i][j][r] * alpha + cb;
                C[(size_t)row * ldc + col] = (TOUT)v;
            }
        }
    }
}

extern "C" void kernel_launch(void* const* d_in, const int* in_sizes, int n_in,
                              void* d_out, int out_size, void* d_ws, size_t ws_size,
                              hipStream_t stream)
{
    (void)in_sizes; (void)n_in; (void)out_size; (void)ws_size;

    const float* rna  = (const float*)d_in[0];  // [8,1024,1280]
    const float* prot = (const float*)d_in[1];  // [8,1024,1344]
    const float* Wq   = (const float*)d_in[2];  // [512,1280]
    const float* bq   = (const float*)d_in[3];  // [512]
    const float* Wk   = (const float*)d_in[4];  // [512,1344]
    const float* bk   = (const float*)d_in[5];  // [512]
    const float* rel  = (const float*)d_in[6];  // [1,8,1,64] -> flat [512]
    float* out = (float*)d_out;                 // [8,1024,1024]

    __bf16* Qws = (__bf16*)d_ws;                              // 8 MiB
    __bf16* Kws = (__bf16*)((char*)d_ws + (size_t)8192 * 512 * 2);

    const float SCALE = 0.125f;      // 64^-0.5
    const float INVH  = 0.125f;      // 1/H

    // GEMM1: Qp = rna @ Wq^T, fold SCALE/H and (bq*SCALE+rel)/H
    dim3 gq(8192 / BM, 512 / BN, 1);
    wmma_gemm_nt<float, __bf16><<<gq, 256, 0, stream>>>(
        rna, 1280, 0LL, Wq, 1280, 0LL, Qws, 512, 0LL, 1280,
        bq, SCALE * INVH, rel, INVH, SCALE * INVH);

    // GEMM2: Kp = prot @ Wk^T + bk
    dim3 gk(8192 / BM, 512 / BN, 1);
    wmma_gemm_nt<float, __bf16><<<gk, 256, 0, stream>>>(
        prot, 1344, 0LL, Wk, 1344, 0LL, Kws, 512, 0LL, 1344,
        bk, 1.0f, nullptr, 0.0f, 1.0f);

    // GEMM3: out[b] = Qp[b] @ Kp[b]^T, fp32 out (async-to-LDS staging)
    dim3 gl(1024 / BM, 1024 / BN, 8);
    wmma_gemm_nt<__bf16, float><<<gl, 256, 0, stream>>>(
        Qws, 512, 1024LL * 512, Kws, 512, 1024LL * 512,
        out, 1024, 1024LL * 1024, 512,
        nullptr, 0.0f, nullptr, 0.0f, 1.0f);
}
